// SingleHeadAttentionInference_3195455668383
// MI455X (gfx1250) — compile-verified, matrix operand order checked by reading, not run on hardware
//
#include <hip/hip_runtime.h>

typedef __attribute__((ext_vector_type(8)))  float   v8f;
typedef __attribute__((ext_vector_type(16))) __bf16  v16bf;
typedef __attribute__((ext_vector_type(8)))  __bf16  v8bf;

#define B_DIM 4
#define S_LEN 4096
#define D_DIM 1024

// ---------------------------------------------------------------- helpers

// Build a 16-element bf16 A/B fragment for v_wmma_f32_16x16x32_bf16.
// Per ISA 7.12.2 (16-bit A 16x32): a lane holds K = {base..base+7, base+16..base+23},
// base = (lane/16)*8. Caller passes a pointer already offset by (lane/16)*8.
__device__ __forceinline__ v16bf frag_ld(const __bf16* base) {
  v8bf lo = *(const v8bf*)(base);
  v8bf hi = *(const v8bf*)(base + 16);
  return __builtin_shufflevector(lo, hi, 0, 1, 2, 3, 4, 5, 6, 7,
                                          8, 9, 10, 11, 12, 13, 14, 15);
}

// LDS byte address of a generic pointer to __shared__ (aperture maps low 32 bits).
__device__ __forceinline__ unsigned lds_addr(const void* p) {
  return (unsigned)(unsigned long long)p;
}

// CDNA5 async global->LDS copy, 16B per lane. Tracked by ASYNCcnt.
__device__ __forceinline__ void async_copy_b128(unsigned lds, const void* gaddr) {
  asm volatile("global_load_async_to_lds_b128 %0, %1, off"
               :: "v"(lds), "v"((unsigned long long)gaddr)
               : "memory");
}

__device__ __forceinline__ void wait_async_le6() {
  asm volatile("s_wait_asynccnt 6" ::: "memory");
}
__device__ __forceinline__ void wait_async_0() {
  asm volatile("s_wait_asynccnt 0" ::: "memory");
}

// ---------------------------------------------------------------- casts
__global__ void cast_f32_bf16(const float* __restrict__ in,
                              __bf16* __restrict__ out, int n) {
  int i = blockIdx.x * blockDim.x + threadIdx.x;
  if (i < n) out[i] = (__bf16)in[i];
}

// ---------------------------------------------------------------- GEMM: C = A @ W^T + bias
// A: [M, 1024] bf16 rm.  W: [1024, 1024] bf16 rm ([N,K], torch Linear layout ->
// contiguous per-lane B fragments, no transpose). Block tile 128x64, K-step 64,
// double-buffered LDS filled by async global->LDS DMA overlapped with WMMA.
#define GM 128
#define GN 64
#define GK 64
#define KTILES (D_DIM / GK)  // 16

__global__ __launch_bounds__(256) void gemm_xWT_bias(
    const __bf16* __restrict__ A, const __bf16* __restrict__ W,
    const float* __restrict__ bias, void* __restrict__ Cout, int f32out) {
  __shared__ __bf16 sA[2][GM * GK];  // 2 x 16KB
  __shared__ __bf16 sW[2][GN * GK];  // 2 x 8KB
  const int tid = threadIdx.x;
  const int wv = tid >> 5;
  const int lane = tid & 31;
  const int lr = lane & 15;  // row (A) / col (B,C)
  const int lh = lane >> 4;  // half: K-group / M-group select
  const int m0 = blockIdx.y * GM;
  const int n0 = blockIdx.x * GN;

  // Staging: thread owns k-chunk (tid&7)*8, base row tid>>3 (+32 per chunk).
  const int srow = tid >> 3;       // 0..31
  const int scol = (tid & 7) * 8;  // 0,8,..,56
  const __bf16* gA = A + (size_t)(m0 + srow) * D_DIM + scol;
  const __bf16* gW = W + (size_t)(n0 + srow) * D_DIM + scol;

  auto stage = [&](int buf, int k0) {
#pragma unroll
    for (int i = 0; i < 4; ++i)  // 128 rows of A
      async_copy_b128(lds_addr(&sA[buf][(srow + 32 * i) * GK + scol]),
                      gA + (size_t)(32 * i) * D_DIM + k0);
#pragma unroll
    for (int i = 0; i < 2; ++i)  // 64 rows of W
      async_copy_b128(lds_addr(&sW[buf][(srow + 32 * i) * GK + scol]),
                      gW + (size_t)(32 * i) * D_DIM + k0);
  };

  v8f acc[4] = {};
  stage(0, 0);
  for (int kt = 0; kt < KTILES; ++kt) {
    const int buf = kt & 1;
    if (kt + 1 < KTILES) {
      stage(buf ^ 1, (kt + 1) * GK);  // kick next tile's DMA first
      wait_async_le6();               // previous 6 ops (this tile) are done
    } else {
      wait_async_0();
    }
    __syncthreads();
#pragma unroll
    for (int kk = 0; kk < GK; kk += 32) {
      v16bf af  = frag_ld(&sA[buf][(wv * 16 + lr) * GK + kk + lh * 8]);
      v16bf wf0 = frag_ld(&sW[buf][(0 * 16 + lr) * GK + kk + lh * 8]);
      v16bf wf1 = frag_ld(&sW[buf][(1 * 16 + lr) * GK + kk + lh * 8]);
      v16bf wf2 = frag_ld(&sW[buf][(2 * 16 + lr) * GK + kk + lh * 8]);
      v16bf wf3 = frag_ld(&sW[buf][(3 * 16 + lr) * GK + kk + lh * 8]);
      acc[0] = __builtin_amdgcn_wmma_f32_16x16x32_bf16(false, af, false, wf0,
                                                       (short)0, acc[0], false, false);
      acc[1] = __builtin_amdgcn_wmma_f32_16x16x32_bf16(false, af, false, wf1,
                                                       (short)0, acc[1], false, false);
      acc[2] = __builtin_amdgcn_wmma_f32_16x16x32_bf16(false, af, false, wf2,
                                                       (short)0, acc[2], false, false);
      acc[3] = __builtin_amdgcn_wmma_f32_16x16x32_bf16(false, af, false, wf3,
                                                       (short)0, acc[3], false, false);
    }
    __syncthreads();
  }

  // C/D layout: lane holds col N=lr, rows M = lh*8 + i in acc element i.
#pragma unroll
  for (int t = 0; t < 4; ++t) {
    int n = n0 + t * 16 + lr;
    float bv = bias[n];
#pragma unroll
    for (int i = 0; i < 8; ++i) {
      int m = m0 + wv * 16 + lh * 8 + i;
      float val = acc[t][i] + bv;
      if (f32out)
        ((float*)Cout)[(size_t)m * D_DIM + n] = val;
      else
        ((__bf16*)Cout)[(size_t)m * D_DIM + n] = (__bf16)val;
    }
  }
}

// ---------------------------------------------------------------- transpose v -> vT
__global__ __launch_bounds__(256) void transpose_bf16(
    const __bf16* __restrict__ v, __bf16* __restrict__ vT) {
  __shared__ __bf16 tile[32][33];
  const int b = blockIdx.z;
  const int s0 = blockIdx.x * 32;
  const int d0 = blockIdx.y * 32;
  const int tx = threadIdx.x & 31;
  const int ty = threadIdx.x >> 5;
  for (int i = ty; i < 32; i += 8)
    tile[i][tx] = v[((size_t)b * S_LEN + s0 + i) * D_DIM + d0 + tx];
  __syncthreads();
  for (int i = ty; i < 32; i += 8)
    vT[((size_t)b * D_DIM + d0 + i) * S_LEN + s0 + tx] = tile[tx][i];
}

// ---------------------------------------------------------------- fused attention
// One workgroup per (batch, 16-query block). 256 threads = 8 waves.
// LDS: Q block 16x1024 bf16 (32KB) + prob buffer 16x4096 bf16 (128KB) + rowsums.
__global__ __launch_bounds__(256) void attention_kernel(
    const __bf16* __restrict__ q, const __bf16* __restrict__ k,
    const __bf16* __restrict__ vT, __bf16* __restrict__ ctx, float scale) {
  extern __shared__ char smem[];
  __bf16* sQ = (__bf16*)smem;                                  // 16 x D_DIM
  __bf16* sP = (__bf16*)(smem + 16 * D_DIM * sizeof(__bf16));  // 16 x S_LEN
  float* sSum = (float*)(smem + (16 * D_DIM + 16 * S_LEN) * sizeof(__bf16));

  const int tid = threadIdx.x;
  const int wv = tid >> 5;
  const int lane = tid & 31;
  const int lr = lane & 15;
  const int lh = lane >> 4;
  const int b = blockIdx.y;
  const int q0 = blockIdx.x * 16;

  // ---- stage Q block into LDS via async DMA (8 x b128 per thread) ----
  const size_t qbase = ((size_t)b * S_LEN + q0) * D_DIM;
#pragma unroll
  for (int i = 0; i < 8; ++i) {
    int c = tid + 256 * i;
    async_copy_b128(lds_addr(&sQ[c * 8]), &q[qbase + (size_t)c * 8]);
  }
  wait_async_0();
  __syncthreads();

  // ---- scores: S = (Q @ K^T) * scale, bf16 into sP ----
  const __bf16* kb = k + (size_t)b * S_LEN * D_DIM;
  for (int nc = wv; nc < S_LEN / 16; nc += 8) {
    const int n0 = nc * 16;
    const __bf16* krow = kb + (size_t)(n0 + lr) * D_DIM + lh * 8;
    if (nc + 8 < S_LEN / 16)  // prefetch next chunk's K row
      __builtin_prefetch(kb + (size_t)(n0 + 128 + lr) * D_DIM + lh * 8, 0, 3);
    v8f acc = {};
#pragma unroll 4
    for (int kk = 0; kk < D_DIM; kk += 32) {
      v16bf af = frag_ld(&sQ[lr * D_DIM + kk + lh * 8]);
      v16bf bf = frag_ld(krow + kk);
      acc = __builtin_amdgcn_wmma_f32_16x16x32_bf16(false, af, false, bf,
                                                    (short)0, acc, false, false);
    }
#pragma unroll
    for (int i = 0; i < 8; ++i) {
      int m = lh * 8 + i;
      sP[m * S_LEN + n0 + lr] = (__bf16)(acc[i] * scale);
    }
  }
  __syncthreads();

  // ---- softmax per row (one wave32 per row, 2 rows/wave) ----
  for (int r = wv; r < 16; r += 8) {
    float mx = -3.0e38f;
    for (int c = lane; c < S_LEN; c += 32)
      mx = fmaxf(mx, (float)sP[r * S_LEN + c]);
#pragma unroll
    for (int off = 16; off > 0; off >>= 1) mx = fmaxf(mx, __shfl_xor(mx, off, 32));
    float sum = 0.f;
    for (int c = lane; c < S_LEN; c += 32) {
      float p = __expf((float)sP[r * S_LEN + c] - mx);
      sP[r * S_LEN + c] = (__bf16)p;
      sum += p;
    }
#pragma unroll
    for (int off = 16; off > 0; off >>= 1) sum += __shfl_xor(sum, off, 32);
    if (lane == 0) sSum[r] = sum;
  }
  __syncthreads();

  // ---- context = P @ V : A = sP (16x4096), B from vT [D][S] (contiguous) ----
  const __bf16* vtb = vT + (size_t)b * D_DIM * S_LEN;
  for (int j = 0; j < 8; ++j) {
    const int n0 = (wv + 8 * j) * 16;
    const __bf16* vrow = vtb + (size_t)(n0 + lr) * S_LEN + lh * 8;
    v8f acc = {};
#pragma unroll 2
    for (int kk = 0; kk < S_LEN; kk += 32) {
      v16bf af = frag_ld(&sP[lr * S_LEN + kk + lh * 8]);
      v16bf bf = frag_ld(vrow + kk);
      acc = __builtin_amdgcn_wmma_f32_16x16x32_bf16(false, af, false, bf,
                                                    (short)0, acc, false, false);
    }
#pragma unroll
    for (int i = 0; i < 8; ++i) {
      int m = lh * 8 + i;
      float val = acc[i] / sSum[m];
      ctx[((size_t)b * S_LEN + q0 + m) * D_DIM + n0 + lr] = (__bf16)val;
    }
  }
}

// ---------------------------------------------------------------- launcher
extern "C" void kernel_launch(void* const* d_in, const int* in_sizes, int n_in,
                              void* d_out, int out_size, void* d_ws,
                              size_t ws_size, hipStream_t stream) {
  (void)in_sizes; (void)n_in; (void)out_size; (void)ws_size;
  const float* x  = (const float*)d_in[0];
  const float* Wq = (const float*)d_in[1];
  const float* bq = (const float*)d_in[2];
  const float* Wk = (const float*)d_in[3];
  const float* bk = (const float*)d_in[4];
  const float* Wv = (const float*)d_in[5];
  const float* bv = (const float*)d_in[6];
  const float* Wo = (const float*)d_in[7];
  const float* bo = (const float*)d_in[8];
  float* out = (float*)d_out;

  const int S = S_LEN, D = D_DIM;
  const int MS = B_DIM * S;  // 16384 token rows

  // workspace (bf16): xb | Wq | Wk | Wv | Wo | q | k | v | vT | ctx  (~200MB)
  char* ws = (char*)d_ws;
  __bf16* xb  = (__bf16*)ws; ws += (size_t)MS * D * 2;
  __bf16* Wqb = (__bf16*)ws; ws += (size_t)D * D * 2;
  __bf16* Wkb = (__bf16*)ws; ws += (size_t)D * D * 2;
  __bf16* Wvb = (__bf16*)ws; ws += (size_t)D * D * 2;
  __bf16* Wob = (__bf16*)ws; ws += (size_t)D * D * 2;
  __bf16* qb  = (__bf16*)ws; ws += (size_t)MS * D * 2;
  __bf16* kbf = (__bf16*)ws; ws += (size_t)MS * D * 2;
  __bf16* vb  = (__bf16*)ws; ws += (size_t)MS * D * 2;
  __bf16* vTb = (__bf16*)ws; ws += (size_t)MS * D * 2;
  __bf16* ctx = (__bf16*)ws; ws += (size_t)MS * D * 2;

  {
    int n = MS * D;
    cast_f32_bf16<<<n / 256, 256, 0, stream>>>(x, xb, n);
    int nw = D * D;
    cast_f32_bf16<<<nw / 256, 256, 0, stream>>>(Wq, Wqb, nw);
    cast_f32_bf16<<<nw / 256, 256, 0, stream>>>(Wk, Wkb, nw);
    cast_f32_bf16<<<nw / 256, 256, 0, stream>>>(Wv, Wvb, nw);
    cast_f32_bf16<<<nw / 256, 256, 0, stream>>>(Wo, Wob, nw);
  }

  dim3 gg(D / GN, MS / GM);  // (16, 128)
  gemm_xWT_bias<<<gg, 256, 0, stream>>>(xb, Wqb, bq, qb, 0);
  gemm_xWT_bias<<<gg, 256, 0, stream>>>(xb, Wkb, bk, kbf, 0);
  gemm_xWT_bias<<<gg, 256, 0, stream>>>(xb, Wvb, bv, vb, 0);

  dim3 tg(S / 32, D / 32, B_DIM);
  transpose_bf16<<<tg, 256, 0, stream>>>(vb, vTb);

  size_t smem = (size_t)(16 * D + 16 * S) * sizeof(__bf16) + 16 * sizeof(float);
  dim3 ag(S / 16, B_DIM);
  attention_kernel<<<ag, 256, smem, stream>>>(qb, kbf, vTb, ctx, 1.0f / 32.0f);

  gemm_xWT_bias<<<gg, 256, 0, stream>>>(ctx, Wob, bo, out, 1);
}
